// QuantumProcessingLayer_68006512165050
// MI455X (gfx1250) — matrix operand order repaired and checked by reading, not compiled, not run on hardware
//
#include <hip/hip_runtime.h>
#include <hip/hip_bf16.h>

typedef __attribute__((ext_vector_type(16))) __bf16 v16bf;
typedef __attribute__((ext_vector_type(8)))  __bf16 v8bf;
typedef __attribute__((ext_vector_type(8)))  float  v8f;
typedef __attribute__((ext_vector_type(4)))  float  f32x4;

#define TWO_PI_F 6.283185307179586f
#define BM 128
#define BN 128
#define BK 32   // x-columns per k-step; one bf16 WMMA K=32 per half (q / q^2)

// gfx1250 native transcendental tanh (single TRANS op, co-executes with WMMA)
#if __has_builtin(__builtin_amdgcn_tanhf)
__device__ __forceinline__ float fast_tanh(float v) { return __builtin_amdgcn_tanhf(v); }
#else
__device__ __forceinline__ float fast_tanh(float v) { return tanhf(v); }
#endif

// Fused: activation(x) -> bf16 feats tiles in LDS -> WMMA bf16 GEMM -> tanh epilogue.
// Software-pipelined: next tile's global loads issue before the WMMA block; the
// activation math for the next tile runs after the WMMAs are issued (VALU/TRANS
// co-executing with the XDL pipe).
__global__ __launch_bounds__(256)
void qpl_fused_wmma_gemm(const float* __restrict__ x,
                         const float* __restrict__ W,
                         const float* __restrict__ bias,
                         float* __restrict__ out,
                         int M, int F, int U)
{
    // LDS (bf16 elements):
    //   Aq  [8 mtiles][32 lanes][16]   = 4096   (fragment-swizzled activation q)
    //   Aq2 [8 mtiles][32 lanes][16]   = 4096   (fragment-swizzled q^2)
    //   Bt  [128 cols][32 k]           = 4096   (W_top, column-major)
    //   Bb  [128 cols][32 k]           = 4096   (W_bot, column-major)
    __shared__ __attribute__((aligned(32))) __bf16 lds[4 * 4096];
    __bf16* Aq  = lds;
    __bf16* Aq2 = lds + 4096;
    __bf16* Bt  = lds + 8192;
    __bf16* Bb  = lds + 12288;

    const int tid   = threadIdx.x;
    const int lane  = tid & 31;        // wave32
    const int wave  = tid >> 5;        // 0..7
    const int waveM = wave >> 1;       // 0..3 -> 32-row band
    const int waveN = wave & 1;        // 0..1 -> 64-col band

    const int mBase = blockIdx.y * BM;
    const int nBase = blockIdx.x * BN;

    // --- activation loader mapping: thread -> (row, 16-col half) of x tile ---
    const int ar    = tid >> 1;        // 0..127 : row in tile
    const int ah    = tid & 1;         // 0/1    : k-half (16 cols)
    const int amt   = ar >> 4;         // m-tile 0..7
    const int alane = ar & 15;         // lane within fragment (low half)
    // A-fragment swizzle: k 0..15 -> lane 'alane'; k 16..31 -> lane+16.
    // element offset within lane = k%16  => contiguous 8-elt chunks.
    __bf16* aq_c0  = Aq  + amt * 512 + (alane + 16 * ah) * 16;
    __bf16* aq_c1  = Aq  + amt * 512 + (alane + 16 * ah) * 16 + 8;
    __bf16* aq2_c0 = Aq2 + amt * 512 + (alane + 16 * ah) * 16;
    __bf16* aq2_c1 = Aq2 + amt * 512 + (alane + 16 * ah) * 16 + 8;

    // --- W loader mapping: thread handles k-pair (2 rows) x 8 columns ---
    const int kp = tid & 15;           // k = 2*kp, 2*kp+1
    const int cg = tid >> 4;           // col group: cols cg*8 .. cg*8+7

    const float* xrow  = x + (size_t)(mBase + ar) * F + ah * 16;            // + kx
    const float* wtrow = W + (size_t)(2 * kp) * U + nBase + cg * 8;         // + kx*U
    const float* wbrow = W + (size_t)(F + 2 * kp) * U + nBase + cg * 8;     // + kx*U

    // -------- pipeline registers --------
    f32x4 xr[4];           // raw x tile slice (16 floats)
    f32x4 wr[8];           // raw W_top/W_bot slices (32 floats)
    v8bf q0, q1, e0, e1;   // bf16 activation fragments, ready for LDS
    unsigned pt[8], pb[8]; // packed bf16 W pairs, ready for LDS

    auto g_load = [&](int kx) {
        const float* xp = xrow + kx;
        xr[0] = *(const f32x4*)(xp + 0);
        xr[1] = *(const f32x4*)(xp + 4);
        xr[2] = *(const f32x4*)(xp + 8);
        xr[3] = *(const f32x4*)(xp + 12);
        const float* wt = wtrow + (size_t)kx * U;
        const float* wb = wbrow + (size_t)kx * U;
        wr[0] = *(const f32x4*)(wt);         wr[1] = *(const f32x4*)(wt + 4);
        wr[2] = *(const f32x4*)(wt + U);     wr[3] = *(const f32x4*)(wt + U + 4);
        wr[4] = *(const f32x4*)(wb);         wr[5] = *(const f32x4*)(wb + 4);
        wr[6] = *(const f32x4*)(wb + U);     wr[7] = *(const f32x4*)(wb + U + 4);
    };

    auto activate = [&]() {
#pragma unroll
        for (int i = 0; i < 16; ++i) {
            float xv = xr[i >> 2][i & 3];
            float qp = xv * TWO_PI_F;
            float qe = fast_tanh(qp) + 0.1f * (__sinf(qp) * __cosf(0.5f * qp));
            __bf16 qb = (__bf16)qe;
            __bf16 eb = (__bf16)(qe * qe);
            if (i < 8) { q0[i] = qb;     e0[i] = eb; }
            else       { q1[i - 8] = qb; e1[i - 8] = eb; }
        }
#pragma unroll
        for (int j = 0; j < 8; ++j) {
            float vt0 = (j < 4) ? wr[0][j] : wr[1][j - 4];   // top, k row
            float vt1 = (j < 4) ? wr[2][j] : wr[3][j - 4];   // top, k+1 row
            float vb0 = (j < 4) ? wr[4][j] : wr[5][j - 4];   // bot, k row
            float vb1 = (j < 4) ? wr[6][j] : wr[7][j - 4];   // bot, k+1 row
            union { __bf16 h[2]; unsigned u; } a, b;
            a.h[0] = (__bf16)vt0; a.h[1] = (__bf16)vt1;
            b.h[0] = (__bf16)vb0; b.h[1] = (__bf16)vb1;
            pt[j] = a.u; pb[j] = b.u;
        }
    };

    auto lds_store = [&]() {
        *(v8bf*)aq_c0  = q0;   // k = half*16 + 0..7
        *(v8bf*)aq_c1  = q1;   // k = half*16 + 8..15
        *(v8bf*)aq2_c0 = e0;
        *(v8bf*)aq2_c1 = e1;
#pragma unroll
        for (int j = 0; j < 8; ++j) {
            const int col = cg * 8 + j;
            *(unsigned*)(Bt + col * 32 + 2 * kp) = pt[j];  // 4B-aligned
            *(unsigned*)(Bb + col * 32 + 2 * kp) = pb[j];
        }
    };

    v8f acc[2][4];
#pragma unroll
    for (int i = 0; i < 2; ++i)
#pragma unroll
        for (int j = 0; j < 4; ++j)
            acc[i][j] = {};

    const int half = lane >> 4;              // B-fragment K half
    const int cLn  = lane & 15;              // B column within tile

    // -------- prologue: fill pipeline for kx = 0 --------
    g_load(0);
    activate();

    for (int kx = 0; kx < F; kx += BK) {
        lds_store();
        __syncthreads();

        const bool more = (kx + BK) < F;
        if (more) g_load(kx + BK);           // async loads in flight during WMMA

        // ================= WMMA compute ==================================
#pragma unroll
        for (int mi = 0; mi < 2; ++mi) {
            const int mt = waveM * 2 + mi;
            v16bf afq = *(const v16bf*)(Aq  + mt * 512 + lane * 16);
            v16bf af2 = *(const v16bf*)(Aq2 + mt * 512 + lane * 16);
#pragma unroll
            for (int ni = 0; ni < 4; ++ni) {
                const int c = (waveN * 4 + ni) * 16 + cLn;
                v16bf bft = *(const v16bf*)(Bt + c * 32 + half * 16);
                v16bf bfb = *(const v16bf*)(Bb + c * 32 + half * 16);
                acc[mi][ni] = __builtin_amdgcn_wmma_f32_16x16x32_bf16(
                    false, afq, false, bft, (short)0, acc[mi][ni], false, false);
                acc[mi][ni] = __builtin_amdgcn_wmma_f32_16x16x32_bf16(
                    false, af2, false, bfb, (short)0, acc[mi][ni], false, false);
            }
        }

        if (more) activate();                // VALU/TRANS co-executes with XDL drain
        __syncthreads();
    }

    // ================= epilogue: tanh(acc + bias) -> out ======================
    // C/D layout: VGPR g, lanes 0-15 -> M=g, N=lane ; lanes 16-31 -> M=8+g, N=lane-16
#pragma unroll
    for (int mi = 0; mi < 2; ++mi) {
        const int row0 = mBase + waveM * 32 + mi * 16 + 8 * (lane >> 4);
#pragma unroll
        for (int ni = 0; ni < 4; ++ni) {
            const int col = nBase + (waveN * 4 + ni) * 16 + (lane & 15);
            const float bv = bias[col];
            float* op = out + (size_t)row0 * U + col;
#pragma unroll
            for (int g = 0; g < 8; ++g) {
                op[(size_t)g * U] = fast_tanh(acc[mi][ni][g] + bv);
            }
        }
    }
}

extern "C" void kernel_launch(void* const* d_in, const int* in_sizes, int n_in,
                              void* d_out, int out_size, void* d_ws, size_t ws_size,
                              hipStream_t stream) {
    const float* x  = (const float*)d_in[0];   // [B,S,F] = [M,F]
    const float* W  = (const float*)d_in[1];   // [2F,U]
    const float* b  = (const float*)d_in[2];   // [U]
    float* out      = (float*)d_out;           // [M,U]

    const int U    = in_sizes[2];              // 2048
    const int twoF = in_sizes[1] / U;          // 4096
    const int F    = twoF / 2;                 // 2048
    const int M    = in_sizes[0] / F;          // 16384

    dim3 grid(U / BN, M / BM);                 // 16 x 128 = 2048 workgroups
    qpl_fused_wmma_gemm<<<grid, 256, 0, stream>>>(x, W, b, out, M, F, U);
}
